// CPINet_36850819400255
// MI455X (gfx1250) — matrix operand order repaired and checked by reading, not compiled, not run on hardware
//
#include <hip/hip_runtime.h>

// ---------------- CDNA5 WMMA plumbing (wave32, gfx1250) ----------------
typedef __attribute__((ext_vector_type(16))) _Float16 v16h;
typedef __attribute__((ext_vector_type(8)))  _Float16 v8h;
typedef __attribute__((ext_vector_type(8)))  float    v8f;
typedef __attribute__((ext_vector_type(4)))  unsigned int v4u;
typedef __attribute__((ext_vector_type(8)))  int      v8i;
typedef __attribute__((ext_vector_type(4)))  int      v4i;

#if __has_builtin(__builtin_amdgcn_tensor_load_to_lds)
#define HAVE_TDM 1
#else
#define HAVE_TDM 0
#endif

__device__ __forceinline__ v16h join16(v8h lo, v8h hi) {
  return __builtin_shufflevector(lo, hi, 0,1,2,3,4,5,6,7,8,9,10,11,12,13,14,15);
}
__device__ __forceinline__ v8f wmma_f16(v16h a, v16h b, v8f c) {
  // D(f32 16x16) = A(f16 16x32) * B(f16 32x16) + C
  return __builtin_amdgcn_wmma_f32_16x16x32_f16(false, a, false, b, (short)0, c, false, false);
}

constexpr int BB = 256, NN = 128, LL = 2048, DD = 64;
constexpr int CW = 23;            // conv kernel width (2*WINDOW+1)
constexpr int SROW = 144;         // conv stripe LDS row stride (f16 elements)
constexpr float EPSF = 1e-6f;

// ---------------- gather kernels ----------------
__global__ void k_gather_atoms(const int* __restrict__ atoms,
                               const float* __restrict__ emb_fp,
                               float* __restrict__ xs) {
  int idx = blockIdx.x * 256 + threadIdx.x;       // over B*N*D
  int d = idx & (DD - 1);
  int bn = idx / DD;
  xs[idx] = emb_fp[atoms[bn] * DD + d];
}

__global__ void k_gather_amino(const int* __restrict__ amino,
                               const float* __restrict__ emb_word,
                               _Float16* __restrict__ ps) {
  long idx = (long)blockIdx.x * 256 + threadIdx.x; // over B*L*D
  int d = (int)(idx & (DD - 1));
  long bl = idx / DD;
  ps[idx] = (_Float16)emb_word[amino[bl] * DD + d];
}

// ---------------- GNN layer: hs = relu(xs W^T + b); xs += adj @ hs ----------------
__global__ void __launch_bounds__(256)
k_gnn(float* __restrict__ xs, const float* __restrict__ adj,
      const float* __restrict__ Wg, const float* __restrict__ bg) {
  const int b = blockIdx.x, tid = threadIdx.x;
  const int lane = tid & 31, wv = tid >> 5, h = lane >> 4, n16 = lane & 15;

  __shared__ __align__(16) unsigned char smem[58368];
  _Float16* Axs  = (_Float16*)smem;             // phase1: xs f16 [128][64]
  _Float16* Bw   = (_Float16*)(smem + 16384);   // W^T fragments [2][4][32][16]
  _Float16* Aadj = (_Float16*)smem;             // phase2: adj f16 [128][128]
  _Float16* hsT  = (_Float16*)(smem + 40960);   // hs transposed [64][136]

  float* xsb = xs + (size_t)b * NN * DD;
  const float* adjb = adj + (size_t)b * NN * NN;

  for (int i = tid; i < NN * DD; i += 256) Axs[i] = (_Float16)xsb[i];
  for (int i = tid; i < 4096; i += 256) {
    int e = i & 15, ln = (i >> 4) & 31, nt = (i >> 9) & 3, kt = i >> 11;
    // B[k][n] = W[nglob][k], k = h*16+e
    Bw[i] = (_Float16)Wg[(nt * 16 + (ln & 15)) * DD + kt * 32 + (ln >> 4) * 16 + e];
  }
  __syncthreads();

  // hs = relu(xs @ W^T + b); each wave: rows [wv*16, wv*16+16)
  const int m = wv * 16 + n16;
#pragma unroll
  for (int nt = 0; nt < 4; ++nt) {
    v8f c = {};
#pragma unroll
    for (int kt = 0; kt < 2; ++kt) {
      v8h alo = *(const v8h*)(Axs + m * DD + kt * 32 + h * 16);
      v8h ahi = *(const v8h*)(Axs + m * DD + kt * 32 + h * 16 + 8);
      const _Float16* bp = Bw + ((kt * 4 + nt) * 32 + lane) * 16;
      c = wmma_f16(join16(alo, ahi), join16(*(const v8h*)bp, *(const v8h*)(bp + 8)), c);
    }
    float bias = bg[nt * 16 + n16];
    v8h hv;
#pragma unroll
    for (int r = 0; r < 8; ++r) hv[r] = (_Float16)fmaxf(c[r] + bias, 0.f);
    *(v8h*)(hsT + (nt * 16 + n16) * 136 + wv * 16 + h * 8) = hv;   // transposed store
  }
  __syncthreads();

  for (int i = tid; i < NN * NN; i += 256) Aadj[i] = (_Float16)adjb[i];
  __syncthreads();

  // xs += adj @ hs
#pragma unroll
  for (int nt = 0; nt < 4; ++nt) {
    v8f c = {};
#pragma unroll
    for (int kt = 0; kt < 4; ++kt) {
      v8h alo = *(const v8h*)(Aadj + m * NN + kt * 32 + h * 16);
      v8h ahi = *(const v8h*)(Aadj + m * NN + kt * 32 + h * 16 + 8);
      const _Float16* bp = hsT + (nt * 16 + n16) * 136 + kt * 32 + h * 16;
      c = wmma_f16(join16(alo, ahi), join16(*(const v8h*)bp, *(const v8h*)(bp + 8)), c);
    }
#pragma unroll
    for (int r = 0; r < 8; ++r) {
      int row = wv * 16 + h * 8 + r, col = nt * 16 + n16;
      xsb[row * DD + col] += c[r];
    }
  }
}

// ---------------- compound masked mean over atoms ----------------
__global__ void k_compound(const float* __restrict__ xs, const float* __restrict__ amask,
                           float* __restrict__ compound) {
  int b = blockIdx.x, d = threadIdx.x;
  float acc = 0.f, msum = 0.f;
  for (int n = 0; n < NN; ++n) {
    float mk = amask[b * NN + n];
    acc += xs[((size_t)b * NN + n) * DD + d] * mk;
    msum += mk;
  }
  compound[b * DD + d] = acc / (msum + EPSF);
}

// ---------------- 23x23 single-channel conv via Toeplitz WMMA ----------------
// out[l,d] = relu(bias + sum_{i,j} in[l+i-11, d+11-j] * k[i,j])
// Input halo stripe staged into LDS by the Tensor Data Mover (TDM):
//   tile 150x128 f16, tensor width 64 -> right edge zero-filled by TDM OOB reads,
//   LDS padded to 144-element rows via pad_interval=64DW / pad_amount=8DW.
__global__ void __launch_bounds__(256)
k_conv(const _Float16* __restrict__ in, _Float16* __restrict__ out,
       const float* __restrict__ kern_g, const float* __restrict__ bias_g, int layer) {
  const int tid = threadIdx.x, lane = tid & 31, wv = tid >> 5, h = lane >> 4, n16 = lane & 15;
  const int b = blockIdx.y;
  const int lbase = blockIdx.x * 128;

  __shared__ __align__(16) _Float16 stripe[150 * SROW];  // rows l-11..l+138, cols c=-16..127
  __shared__ float kern[CW * CW];

  const _Float16* inb = in + (size_t)b * LL * DD;
  for (int i = tid; i < CW * CW; i += 256) kern[i] = kern_g[layer * CW * CW + i];

  int r0 = lbase - 11;
  int topZero = 0;
  if (r0 < 0) { topZero = -r0; r0 = 0; }

#if HAVE_TDM
  {
    // Manually zero only the 16 left-pad columns per row (TDM pad skips them)
    // and the top-clamped rows of the first l-block.
    v8h z = {};
    for (int i = tid; i < 150; i += 256) {
      *(v8h*)(stripe + i * SROW) = z;
      *(v8h*)(stripe + i * SROW + 8) = z;
    }
    for (int i = tid; i < topZero * SROW; i += 256) stripe[i] = (_Float16)0.f;
    if (tid < 32) {  // one wave issues the DMA descriptor
      unsigned lds_off =
          (unsigned)(unsigned long long)(stripe + topZero * SROW + 16);
      unsigned long long ga = (unsigned long long)(inb + (size_t)r0 * DD);
      int td1 = LL - r0;           // rows available; OOB rows read as zero
      int tile1 = 150 - topZero;   // rows to move
      // D# group0: count=1 | lds_addr | global_addr(57b) | type=2
      v4u g0 = { 1u, lds_off, (unsigned)ga,
                 (unsigned)((ga >> 32) & 0x1FFFFFFull) | 0x80000000u };
      // D# group1: data_size=2B(16), pad_enable(20), pad_interval=5(64DW),
      //            pad_amount=7(8DW); tensor_dim0=64, tensor_dim1=td1,
      //            tile_dim0=128, tile_dim1=tile1, tensor_dim0_stride=64
      v8i g1 = { (int)0x0F510000u,
                 (int)(64u << 16),
                 (int)(((unsigned)td1 & 0xFFFFu) << 16),
                 (int)(((unsigned)td1 >> 16) | (128u << 16)),
                 (int)((unsigned)tile1 & 0xFFFFu),
                 64, 0, 0 };
      v4i gz = { 0, 0, 0, 0 };
#if defined(__clang_major__) && (__clang_major__ >= 23)
      v8i gz8 = { 0, 0, 0, 0, 0, 0, 0, 0 };
      __builtin_amdgcn_tensor_load_to_lds(g0, g1, gz, gz, gz8, 0);
#else
      __builtin_amdgcn_tensor_load_to_lds(g0, g1, gz, gz, 0);
#endif
      __builtin_amdgcn_s_wait_tensorcnt(0);
    }
  }
#else
  for (int i = tid; i < 150 * SROW; i += 256) {
    int r = i / SROW, c = i % SROW;
    int gl = lbase + r - 11, gd = c - 16;
    _Float16 v = (_Float16)0.f;
    if (gl >= 0 && gl < LL && gd >= 0 && gd < DD) v = inb[gl * DD + gd];
    stripe[i] = v;
  }
#endif
  __syncthreads();

  v8f C[4] = {{}, {}, {}, {}};
  for (int dy = 0; dy < CW; ++dy) {
    // Banded Toeplitz B fragments (shared by all d-tiles):
    // chunk0 (c0=n0-16): B0[k][n]=k[dy, n-k+27]; chunk1 (c0=n0+16): B1[k][n]=k[dy, n-k-5]
    v16h B0, B1;
#pragma unroll
    for (int e = 0; e < 16; ++e) {
      int k = h * 16 + e;
      int j0 = n16 - k + 27, j1 = n16 - k - 5;
      B0[e] = (j0 >= 0 && j0 < CW) ? (_Float16)kern[dy * CW + j0] : (_Float16)0.f;
      B1[e] = (j1 >= 0 && j1 < CW) ? (_Float16)kern[dy * CW + j1] : (_Float16)0.f;
    }
    const int arow = wv * 16 + n16 + dy;  // lds row = (l - lbase) + 11 + (dy - 11)
#pragma unroll
    for (int nt = 0; nt < 4; ++nt) {
      const _Float16* base0 = stripe + arow * SROW + nt * 16 + h * 16;  // lds col = n0 + k
      C[nt] = wmma_f16(join16(*(const v8h*)base0, *(const v8h*)(base0 + 8)), B0, C[nt]);
      const _Float16* base1 = base0 + 32;
      C[nt] = wmma_f16(join16(*(const v8h*)base1, *(const v8h*)(base1 + 8)), B1, C[nt]);
    }
  }

  const float bias = bias_g[layer];
  _Float16* outb = out + (size_t)b * LL * DD;
#pragma unroll
  for (int nt = 0; nt < 4; ++nt)
#pragma unroll
    for (int r = 0; r < 8; ++r) {
      int l = lbase + wv * 16 + h * 8 + r;
      outb[l * DD + nt * 16 + n16] = (_Float16)fmaxf(C[nt][r] + bias, 0.f);
    }
}

// ---------------- hs = relu(ps @ W_att^T + b_att) * mask  (M = B*L streaming GEMM) ----------------
__global__ void __launch_bounds__(256)
k_hs_att(const _Float16* __restrict__ ps, const float* __restrict__ W,
         const float* __restrict__ bv, const float* __restrict__ pmask,
         _Float16* __restrict__ hs) {
  const int tid = threadIdx.x, lane = tid & 31, wv = tid >> 5, h = lane >> 4, n16 = lane & 15;
  __shared__ _Float16 Bw[4096];
  for (int i = tid; i < 4096; i += 256) {
    int e = i & 15, ln = (i >> 4) & 31, nt = (i >> 9) & 3, kt = i >> 11;
    Bw[i] = (_Float16)W[(nt * 16 + (ln & 15)) * DD + kt * 32 + (ln >> 4) * 16 + e];
  }
  __syncthreads();

  const size_t rowbase = (size_t)blockIdx.x * 128 + wv * 16;
#pragma unroll
  for (int nt = 0; nt < 4; ++nt) {
    v8f c = {};
#pragma unroll
    for (int kt = 0; kt < 2; ++kt) {
      const _Float16* ap = ps + (rowbase + n16) * DD + kt * 32 + h * 16;
      const _Float16* bp = Bw + ((kt * 4 + nt) * 32 + lane) * 16;
      c = wmma_f16(join16(*(const v8h*)ap, *(const v8h*)(ap + 8)),
                   join16(*(const v8h*)bp, *(const v8h*)(bp + 8)), c);
    }
    float bias = bv[nt * 16 + n16];
#pragma unroll
    for (int r = 0; r < 8; ++r) {
      size_t row = rowbase + h * 8 + r;
      hs[row * DD + nt * 16 + n16] = (_Float16)(fmaxf(c[r] + bias, 0.f) * pmask[row]);
    }
  }
}

// ---------------- small f32 tail kernels ----------------
__global__ void k_h_att(const float* __restrict__ compound, const float* __restrict__ W,
                        const float* __restrict__ bv, float* __restrict__ hout) {
  int idx = blockIdx.x * 256 + threadIdx.x;  // B*D
  int e = idx & (DD - 1), b = idx / DD;
  float acc = bv[e];
  for (int d = 0; d < DD; ++d) acc += compound[b * DD + d] * W[e * DD + d];
  hout[idx] = fmaxf(acc, 0.f);
}

__global__ void k_weights(const float* __restrict__ hatt, const _Float16* __restrict__ hs,
                          float* __restrict__ wts) {
  long idx = (long)blockIdx.x * 256 + threadIdx.x;  // B*L
  long b = idx / LL;
  float acc = 0.f;
  const _Float16* hp = hs + idx * DD;
  const float* ha = hatt + b * DD;
  for (int d = 0; d < DD; ++d) acc += ha[d] * (float)hp[d];
  wts[idx] = tanhf(acc);
}

__global__ void k_protein(const float* __restrict__ wts, const _Float16* __restrict__ hs,
                          const float* __restrict__ pmask, float* __restrict__ protein) {
  int b = blockIdx.x, d = threadIdx.x;
  float acc = 0.f, msum = 0.f;
  for (int l = 0; l < LL; ++l) {
    acc += wts[(size_t)b * LL + l] * (float)hs[((size_t)b * LL + l) * DD + d];
    msum += pmask[(size_t)b * LL + l];
  }
  protein[b * DD + d] = acc / (msum + EPSF);
}

__global__ void k_out(const float* __restrict__ compound, const float* __restrict__ protein,
                      const float* __restrict__ W_out, const float* __restrict__ b_out,
                      const float* __restrict__ W_int, const float* __restrict__ b_int,
                      float* __restrict__ out) {
  int b = blockIdx.x, t = threadIdx.x;  // 128 threads
  __shared__ float cat[128], tmp[128];
  cat[t] = (t < DD) ? compound[b * DD + t] : protein[b * DD + t - DD];
  __syncthreads();
  for (int j = 0; j < 2; ++j) {
    float acc = b_out[j * 128 + t];
    for (int d = 0; d < 128; ++d) acc += cat[d] * W_out[(j * 128 + t) * 128 + d];
    tmp[t] = fmaxf(acc, 0.f);
    __syncthreads();
    cat[t] = tmp[t];
    __syncthreads();
  }
  if (t < 2) {
    float acc = b_int[t];
    for (int d = 0; d < 128; ++d) acc += cat[d] * W_int[t * 128 + d];
    out[b * 2 + t] = acc;
  }
}

// ---------------- launcher ----------------
extern "C" void kernel_launch(void* const* d_in, const int* in_sizes, int n_in,
                              void* d_out, int out_size, void* d_ws, size_t ws_size,
                              hipStream_t stream) {
  const int*   atoms      = (const int*)d_in[0];
  const float* atoms_mask = (const float*)d_in[1];
  const float* adjacency  = (const float*)d_in[2];
  const int*   amino      = (const int*)d_in[3];
  const float* amino_mask = (const float*)d_in[4];
  const float* emb_fp     = (const float*)d_in[5];
  const float* emb_word   = (const float*)d_in[6];
  const float* W_gnn      = (const float*)d_in[7];
  const float* b_gnn      = (const float*)d_in[8];
  const float* conv_k     = (const float*)d_in[9];
  const float* conv_b     = (const float*)d_in[10];
  const float* W_att      = (const float*)d_in[11];
  const float* b_att      = (const float*)d_in[12];
  const float* W_out      = (const float*)d_in[13];
  const float* b_out      = (const float*)d_in[14];
  const float* W_int      = (const float*)d_in[15];
  const float* b_int      = (const float*)d_in[16];
  float* out = (float*)d_out;

  char* p = (char*)d_ws;
  float*     xs       = (float*)p;     p += (size_t)BB * NN * DD * 4;  // 8 MB
  float*     compound = (float*)p;     p += (size_t)BB * DD * 4;
  _Float16*  ps0      = (_Float16*)p;  p += (size_t)BB * LL * DD * 2;  // 64 MB
  _Float16*  ps1      = (_Float16*)p;  p += (size_t)BB * LL * DD * 2;  // 64 MB
  _Float16*  hsA      = (_Float16*)p;  p += (size_t)BB * LL * DD * 2;  // 64 MB
  float*     hatt     = (float*)p;     p += (size_t)BB * DD * 4;
  float*     wts      = (float*)p;     p += (size_t)BB * LL * 4;       // 2 MB
  float*     protein  = (float*)p;     p += (size_t)BB * DD * 4;

  // ---- compound (GNN) branch ----
  k_gather_atoms<<<BB * NN * DD / 256, 256, 0, stream>>>(atoms, emb_fp, xs);
  for (int i = 0; i < 3; ++i)
    k_gnn<<<BB, 256, 0, stream>>>(xs, adjacency, W_gnn + i * DD * DD, b_gnn + i * DD);
  k_compound<<<BB, DD, 0, stream>>>(xs, atoms_mask, compound);

  // ---- protein (CNN) branch ----
  k_gather_amino<<<(int)((size_t)BB * LL * DD / 256), 256, 0, stream>>>(amino, emb_word, ps0);
  dim3 cgrid(LL / 128, BB);
  k_conv<<<cgrid, 256, 0, stream>>>(ps0, ps1, conv_k, conv_b, 0);
  k_conv<<<cgrid, 256, 0, stream>>>(ps1, ps0, conv_k, conv_b, 1);
  k_conv<<<cgrid, 256, 0, stream>>>(ps0, ps1, conv_k, conv_b, 2);

  // ---- attention + pooling ----
  k_hs_att<<<BB * LL / 128, 256, 0, stream>>>(ps1, W_att, b_att, amino_mask, hsA);
  k_h_att<<<BB * DD / 256, 256, 0, stream>>>(compound, W_att, b_att, hatt);
  k_weights<<<BB * LL / 256, 256, 0, stream>>>(hatt, hsA, wts);
  k_protein<<<BB, DD, 0, stream>>>(wts, hsA, amino_mask, protein);

  // ---- output MLP ----
  k_out<<<BB, 128, 0, stream>>>(compound, protein, W_out, b_out, W_int, b_int, out);
}
